// TitanModel_70205535420757
// MI455X (gfx1250) — matrix operand order repaired and checked by reading, not compiled, not run on hardware
//
#include <hip/hip_runtime.h>
#include <cstdint>
#include <cstddef>

#define DEVINL __device__ __forceinline__

typedef __bf16 bf16_t;
typedef __bf16 bf16x8  __attribute__((ext_vector_type(8)));
typedef __bf16 bf16x16 __attribute__((ext_vector_type(16)));
typedef float  f32x8   __attribute__((ext_vector_type(8)));

constexpr int    Bc = 64, Lc = 512, Dc = 384, Hc = 12, DHc = 32, NLc = 6, FFc = 1536, CHc = 64;
constexpr size_t BLc = (size_t)Bc * Lc;   // 32768 rows
constexpr size_t BHc = (size_t)Bc * Hc;   // 768 (b,h) pairs

DEVINL float sigmoidf_(float x) { return 1.f / (1.f + __expf(-x)); }
DEVINL float geluf_(float x) { return 0.5f * x * (1.f + erff(x * 0.7071067811865475f)); }

DEVINL f32x8 fzero8() {
  f32x8 z;
#pragma unroll
  for (int i = 0; i < 8; ++i) z[i] = 0.f;
  return z;
}

DEVINL f32x8 wmma_bf16(bf16x16 a, bf16x16 b, f32x8 c) {
  // D = A(16x32 bf16) x B(32x16 bf16) + C(16x16 f32)
  return __builtin_amdgcn_wmma_f32_16x16x32_bf16(false, a, false, b, (short)0, c, false, false);
}

// Build a v16bf fragment from two contiguous 16-byte chunks.
DEVINL bf16x16 load_frag(const bf16_t* p0, const bf16_t* p1) {
  union { bf16x16 v; bf16x8 h[2]; } u;
  u.h[0] = *(const bf16x8*)p0;
  u.h[1] = *(const bf16x8*)p1;
  return u.v;
}

DEVINL float block_reduce_sum(float v, float* sbuf) {
  __syncthreads();
#pragma unroll
  for (int o = 16; o > 0; o >>= 1) v += __shfl_xor(v, o, 32);
  if ((threadIdx.x & 31) == 0) sbuf[threadIdx.x >> 5] = v;
  __syncthreads();
  if (threadIdx.x == 0) {
    float s = 0.f;
    int nw = (int)(blockDim.x >> 5);
    for (int i = 0; i < nw; ++i) s += sbuf[i];
    sbuf[0] = s;
  }
  __syncthreads();
  return sbuf[0];
}

// ---------------------------------------------------------------------------
// C(M,N) = A(M,K)_bf16 @ W(N,K)_bf16^T (+bias). 128 threads = 4 waves.
// Each wave owns a 64x32 tile (4x2 WMMA tiles) -> 8 WMMA per 6 fragment loads
// per K-step; block covers 128x64.
// A-frag: two 16B loads at K offsets {0,16} (lanes 0-15) / {8,24} (lanes 16-31).
// B-frag: two 16B loads at K offsets {0,16} (lanes 0-15) / {8,24} (lanes 16-31).
// EPI = 0: f32 store (+bias).  EPI = 1: bias + exact GeLU, bf16 store.
// ---------------------------------------------------------------------------
template <int EPI>
__global__ __launch_bounds__(128) void gemm_bf16_kernel(
    const bf16_t* __restrict__ A, const bf16_t* __restrict__ W,
    const float* __restrict__ bias, void* __restrict__ Cout,
    int M, int N, int K) {
  const int wave   = threadIdx.x >> 5;
  const int lane   = threadIdx.x & 31;
  const int laneLo = lane & 15;
  const bool hi    = lane >= 16;
  const int tileM  = blockIdx.x * 128 + (wave >> 1) * 64;
  const int tileN  = blockIdx.y * 64 + (wave & 1) * 32;
  const int aOff   = hi ? 8 : 0;
  const int bOff   = hi ? 8 : 0;

  f32x8 acc[4][2];
#pragma unroll
  for (int mt = 0; mt < 4; ++mt) {
    acc[mt][0] = fzero8();
    acc[mt][1] = fzero8();
  }

  for (int k0 = 0; k0 < K; k0 += 32) {
    bf16x16 af[4], bw[2];
#pragma unroll
    for (int mt = 0; mt < 4; ++mt) {
      const bf16_t* p = A + (size_t)(tileM + mt * 16 + laneLo) * K + k0 + aOff;
      af[mt] = load_frag(p, p + 16);
      if (k0 + 32 < K) __builtin_prefetch(p + 32, 0, 0);  // global_prefetch_b8
    }
#pragma unroll
    for (int nt = 0; nt < 2; ++nt) {
      const bf16_t* p = W + (size_t)(tileN + nt * 16 + laneLo) * K + k0 + bOff;
      bw[nt] = load_frag(p, p + 16);
    }
#pragma unroll
    for (int mt = 0; mt < 4; ++mt)
#pragma unroll
      for (int nt = 0; nt < 2; ++nt)
        acc[mt][nt] = wmma_bf16(af[mt], bw[nt], acc[mt][nt]);
  }

#pragma unroll
  for (int mt = 0; mt < 4; ++mt) {
    const int mBase = tileM + mt * 16 + (hi ? 8 : 0);
#pragma unroll
    for (int nt = 0; nt < 2; ++nt) {
      const int n = tileN + nt * 16 + laneLo;
      const float bv = bias ? bias[n] : 0.f;
      if (EPI == 0) {
        float* C = (float*)Cout;
#pragma unroll
        for (int g = 0; g < 8; ++g)
          C[(size_t)(mBase + g) * N + n] = acc[mt][nt][g] + bv;
      } else {
        bf16_t* C = (bf16_t*)Cout;
#pragma unroll
        for (int g = 0; g < 8; ++g)
          C[(size_t)(mBase + g) * N + n] = (bf16_t)geluf_(acc[mt][nt][g] + bv);
      }
    }
  }
}

// ---------------------------------------------------------------------------
// Chunked delta-rule scan: one wave per (b, h, branch). State S (32x32 f32)
// lives in 4 WMMA accumulators; decays by wfac per 64-row chunk.
//   S += v^T k  :  A = v^T from VT (d-major), B = k from KT (d-major rows = e)
//   o  = q S    :  A = q (l-major),           B = S via LDS transpose [e][d]
// ---------------------------------------------------------------------------
__global__ __launch_bounds__(32) void delta_kernel(
    const bf16_t* __restrict__ QB, const bf16_t* __restrict__ KT,
    const bf16_t* __restrict__ VT, const bf16_t* __restrict__ KTS,
    const bf16_t* __restrict__ VTS, const float* __restrict__ wfac2,
    float* __restrict__ OF, float* __restrict__ OS) {
  __shared__ __align__(16) bf16_t sT[32 * 32];
  const int bh     = blockIdx.x;
  const int branch = blockIdx.y;
  const bf16_t* kt = (branch ? KTS : KT) + (size_t)bh * DHc * Lc;
  const bf16_t* vt = (branch ? VTS : VT) + (size_t)bh * DHc * Lc;
  const bf16_t* qp = QB + (size_t)bh * Lc * DHc;
  const float wfac = wfac2[branch];
  float* out = (branch ? OS : OF) + (size_t)(bh / Hc) * Lc * Dc + (size_t)(bh % Hc) * DHc;

  const int lane   = threadIdx.x;
  const int laneLo = lane & 15;
  const bool hi    = lane >= 16;
  const int aOff   = hi ? 8 : 0;
  const int bOff   = hi ? 8 : 0;

  f32x8 S[2][2];
  S[0][0] = fzero8(); S[0][1] = fzero8();
  S[1][0] = fzero8(); S[1][1] = fzero8();

  for (int c = 0; c < Lc / CHc; ++c) {
    const int l0c = c * CHc;
    // state decay
#pragma unroll
    for (int mt = 0; mt < 2; ++mt)
#pragma unroll
      for (int nt = 0; nt < 2; ++nt)
#pragma unroll
        for (int g = 0; g < 8; ++g) S[mt][nt][g] *= wfac;
    // S += v^T k over this chunk (K = 64 -> two WMMA K-steps)
#pragma unroll
    for (int kk = 0; kk < 2; ++kk) {
      const int l0 = l0c + kk * 32;
      bf16x16 av[2], bk[2];
#pragma unroll
      for (int mt = 0; mt < 2; ++mt) {
        const bf16_t* p = vt + (size_t)(mt * 16 + laneLo) * Lc + l0 + aOff;
        av[mt] = load_frag(p, p + 16);
      }
#pragma unroll
      for (int nt = 0; nt < 2; ++nt) {
        const bf16_t* p = kt + (size_t)(nt * 16 + laneLo) * Lc + l0 + bOff;
        bk[nt] = load_frag(p, p + 16);
      }
#pragma unroll
      for (int mt = 0; mt < 2; ++mt)
#pragma unroll
        for (int nt = 0; nt < 2; ++nt)
          S[mt][nt] = wmma_bf16(av[mt], bk[nt], S[mt][nt]);
    }
    // spill S transposed ([e][d], d contiguous) to LDS as bf16
#pragma unroll
    for (int mt = 0; mt < 2; ++mt) {
      const int dBase = mt * 16 + (hi ? 8 : 0);
#pragma unroll
      for (int nt = 0; nt < 2; ++nt) {
        const int e = nt * 16 + laneLo;
        bf16x8 pk;
#pragma unroll
        for (int g = 0; g < 8; ++g) pk[g] = (bf16_t)S[mt][nt][g];
        *(bf16x8*)(&sT[e * 32 + dBase]) = pk;
      }
    }
    asm volatile("s_wait_dscnt 0" ::: "memory");
    // o_chunk = q_chunk @ S   (K = DHc = 32, single WMMA step)
#pragma unroll
    for (int lt = 0; lt < 4; ++lt) {
      const bf16_t* p = qp + (size_t)(l0c + lt * 16 + laneLo) * DHc + aOff;
      bf16x16 aq = load_frag(p, p + 16);
      const int lBase = l0c + lt * 16 + (hi ? 8 : 0);
#pragma unroll
      for (int et = 0; et < 2; ++et) {
        const bf16_t* sp = &sT[(et * 16 + laneLo) * 32 + bOff];
        bf16x16 bs = load_frag(sp, sp + 16);
        f32x8 o = wmma_bf16(aq, bs, fzero8());
        const int eCol = et * 16 + laneLo;
#pragma unroll
        for (int g = 0; g < 8; ++g)
          out[(size_t)(lBase + g) * Dc + eCol] = o[g];
      }
    }
    asm volatile("s_wait_dscnt 0" ::: "memory"); // WAR guard before next LDS spill
  }
}

// ---------------------------------------------------------------------------
__global__ void cvt_bf16_kernel(const float* __restrict__ src, bf16_t* __restrict__ dst, size_t n) {
  size_t i = (size_t)blockIdx.x * blockDim.x + threadIdx.x;
  if (i < n) dst[i] = (bf16_t)src[i];
}

__global__ __launch_bounds__(128) void wfac_kernel(const float* __restrict__ fdec,
                                                   const float* __restrict__ sdec,
                                                   float* __restrict__ wf) {
  __shared__ float sbuf[8];
  const int layer = blockIdx.x;
  const float* f = fdec + (size_t)layer * Hc * DHc;
  const float* s = sdec + (size_t)layer * Hc * DHc;
  float sf = 0.f, ss = 0.f;
  for (int i = threadIdx.x; i < Hc * DHc; i += blockDim.x) {
    sf += sigmoidf_(f[i]);
    ss += sigmoidf_(s[i]);
  }
  sf = block_reduce_sum(sf, sbuf);
  ss = block_reduce_sum(ss, sbuf);
  if (threadIdx.x == 0) {
    wf[layer * 2 + 0] = powf(sf / (float)(Hc * DHc), (float)CHc);
    wf[layer * 2 + 1] = powf(ss / (float)(Hc * DHc), (float)CHc);
  }
}

__global__ __launch_bounds__(128) void embed_ln_kernel(
    const int* __restrict__ ids, const float* __restrict__ we,
    const float* __restrict__ pe, const float* __restrict__ te,
    const float* __restrict__ lw, const float* __restrict__ lb,
    float* __restrict__ X) {
  __shared__ float sbuf[8];
  const size_t bl = blockIdx.x;
  const int l = (int)(bl % Lc);
  const int id = ids[bl];
  float v[3]; float sum = 0.f;
#pragma unroll
  for (int i = 0; i < 3; ++i) {
    int d = (int)threadIdx.x + i * 128;
    float x = we[(size_t)id * Dc + d] + pe[(size_t)l * Dc + d] + te[d];
    v[i] = x; sum += x;
  }
  float mean = block_reduce_sum(sum, sbuf) * (1.f / Dc);
  float vs = 0.f;
#pragma unroll
  for (int i = 0; i < 3; ++i) { float dd = v[i] - mean; vs += dd * dd; }
  float var = block_reduce_sum(vs, sbuf) * (1.f / Dc);
  float inv = rsqrtf(var + 1e-12f);
#pragma unroll
  for (int i = 0; i < 3; ++i) {
    int d = (int)threadIdx.x + i * 128;
    X[bl * Dc + d] = (v[i] - mean) * inv * lw[d] + lb[d];
  }
}

__global__ __launch_bounds__(128) void residual_ln_kernel(
    float* __restrict__ X, const float* __restrict__ T,
    const float* __restrict__ lw, const float* __restrict__ lb,
    bf16_t* __restrict__ XB) {
  __shared__ float sbuf[8];
  const size_t bl = blockIdx.x;
  float v[3]; float sum = 0.f;
#pragma unroll
  for (int i = 0; i < 3; ++i) {
    int d = (int)threadIdx.x + i * 128;
    float x = X[bl * Dc + d] + T[bl * Dc + d];
    v[i] = x; sum += x;
  }
  float mean = block_reduce_sum(sum, sbuf) * (1.f / Dc);
  float vs = 0.f;
#pragma unroll
  for (int i = 0; i < 3; ++i) { float dd = v[i] - mean; vs += dd * dd; }
  float var = block_reduce_sum(vs, sbuf) * (1.f / Dc);
  float inv = rsqrtf(var + 1e-12f);
#pragma unroll
  for (int i = 0; i < 3; ++i) {
    int d = (int)threadIdx.x + i * 128;
    float y = (v[i] - mean) * inv * lw[d] + lb[d];
    X[bl * Dc + d] = y;
    XB[bl * Dc + d] = (bf16_t)y;
  }
}

__global__ void conv_silu_kernel(const float* __restrict__ X, const float* __restrict__ Wc,
                                 const float* __restrict__ Bb, bf16_t* __restrict__ XB) {
  size_t idx = (size_t)blockIdx.x * blockDim.x + threadIdx.x;
  if (idx >= BLc * Dc) return;
  const int d = (int)(idx % Dc);
  const size_t bl = idx / Dc;
  const int l = (int)(bl % Lc);
  const size_t bbase = (bl - (size_t)l) * Dc;  // (b*L)*D
  float acc = Bb[d];
#pragma unroll
  for (int t = 0; t < 4; ++t) {
    int ls = l + t - 2;
    if (ls >= 0 && ls < Lc)
      acc += X[bbase + (size_t)ls * Dc + d] * Wc[d * 4 + t];
  }
  XB[idx] = (bf16_t)(acc * sigmoidf_(acc));
}

// gating + surprise gate + bf16 repack (q l-major; k/v/k*s/v*s d-major transposed)
__global__ __launch_bounds__(128) void fuse_qkvg_kernel(
    const float* __restrict__ qf, const float* __restrict__ kf,
    const float* __restrict__ vf, const float* __restrict__ gf,
    const float* __restrict__ sW, const float* __restrict__ sB,
    bf16_t* __restrict__ QB, bf16_t* __restrict__ KT, bf16_t* __restrict__ VT,
    bf16_t* __restrict__ KTS, bf16_t* __restrict__ VTS) {
  const size_t gw = ((size_t)blockIdx.x * blockDim.x + threadIdx.x) >> 5;
  const int lane = (int)(threadIdx.x & 31);
  if (gw >= BLc * Hc) return;
  const size_t bl = gw / Hc;
  const int h = (int)(gw % Hc);
  const int b = (int)(bl / Lc);
  const int l = (int)(bl % Lc);
  const size_t base = bl * Dc + (size_t)h * DHc + lane;
  const float q = qf[base], k = kf[base];
  const float vg = vf[base] * sigmoidf_(gf[base]);
  float p = k * sW[lane] + vg * sW[DHc + lane];
#pragma unroll
  for (int o = 16; o > 0; o >>= 1) p += __shfl_xor(p, o, 32);
  const float s = sigmoidf_(10.f * (p + sB[0]));
  const size_t bh = (size_t)b * Hc + h;
  QB[(bh * Lc + l) * DHc + lane] = (bf16_t)q;
  const size_t tb = (bh * DHc + lane) * Lc + l;
  KT[tb]  = (bf16_t)k;
  VT[tb]  = (bf16_t)vg;
  KTS[tb] = (bf16_t)(k * s);
  VTS[tb] = (bf16_t)(vg * s);
}

__global__ void blend_kernel(const float* __restrict__ OF, const float* __restrict__ OS,
                             const float* __restrict__ alphap, bf16_t* __restrict__ OB) {
  size_t i = (size_t)blockIdx.x * blockDim.x + threadIdx.x;
  if (i >= BLc * Dc) return;
  const float a = sigmoidf_(alphap[0]);
  OB[i] = (bf16_t)(a * OF[i] + (1.f - a) * OS[i]);
}

__global__ __launch_bounds__(128) void pool_norm_kernel(
    const float* __restrict__ X, const int* __restrict__ mask, float* __restrict__ out) {
  __shared__ float sbuf[8];
  __shared__ float emb[Dc];
  const int b = blockIdx.x;
  float ms = 0.f;
  for (int l = (int)threadIdx.x; l < Lc; l += (int)blockDim.x) ms += (float)mask[(size_t)b * Lc + l];
  const float denom = fmaxf(block_reduce_sum(ms, sbuf), 1e-9f);
  float ssp = 0.f;
  for (int d = (int)threadIdx.x; d < Dc; d += (int)blockDim.x) {
    float s = 0.f;
    for (int l = 0; l < Lc; ++l)
      s += X[((size_t)b * Lc + l) * Dc + d] * (float)mask[(size_t)b * Lc + l];
    float e = s / denom;
    emb[d] = e;
    ssp += e * e;
  }
  const float ss = block_reduce_sum(ssp, sbuf);
  const float inv = 1.f / fmaxf(sqrtf(ss), 1e-12f);
  for (int d = (int)threadIdx.x; d < Dc; d += (int)blockDim.x)
    out[(size_t)b * Dc + d] = emb[d] * inv;
}

// ---------------------------------------------------------------------------
extern "C" void kernel_launch(void* const* d_in, const int* in_sizes, int n_in,
                              void* d_out, int out_size, void* d_ws, size_t ws_size,
                              hipStream_t stream) {
  (void)in_sizes; (void)n_in; (void)out_size; (void)ws_size;
  const int*   ids   = (const int*)d_in[0];
  const int*   amask = (const int*)d_in[1];
  const float* we    = (const float*)d_in[2];
  const float* pe    = (const float*)d_in[3];
  const float* te    = (const float*)d_in[4];
  const float* elnw  = (const float*)d_in[5];
  const float* elnb  = (const float*)d_in[6];
  const float* qW    = (const float*)d_in[7];
  const float* kW    = (const float*)d_in[8];
  const float* vW    = (const float*)d_in[9];
  const float* oW    = (const float*)d_in[10];
  const float* gW    = (const float*)d_in[11];
  const float* convW = (const float*)d_in[12];
  const float* convB = (const float*)d_in[13];
  const float* fdec  = (const float*)d_in[14];
  const float* sdec  = (const float*)d_in[15];
  const float* surW  = (const float*)d_in[16];
  const float* surB  = (const float*)d_in[17];
  const float* alpha = (const float*)d_in[18];
  const float* ln1w  = (const float*)d_in[19];
  const float* ln1b  = (const float*)d_in[20];
  const float* fW1   = (const float*)d_in[21];
  const float* fB1   = (const float*)d_in[22];
  const float* fW2   = (const float*)d_in[23];
  const float* fB2   = (const float*)d_in[24];
  const float* ln2w  = (const float*)d_in[25];
  const float* ln2b  = (const float*)d_in[26];
  float* out = (float*)d_out;

  char* ws = (char*)d_ws;
  constexpr size_t SZ_XD_F = BLc * Dc * sizeof(float);
  constexpr size_t SZ_XD_B = BLc * Dc * sizeof(uint16_t);
  constexpr size_t SZ_XF_B = BLc * FFc * sizeof(uint16_t);
  size_t off = 0;
  auto take = [&](size_t sz) { size_t o = off; off += (sz + 255) & ~(size_t)255; return o; };
  const size_t oX   = take(SZ_XD_F);
  const size_t oTF  = take(SZ_XD_F);
  const size_t oBIG = take(4 * SZ_XD_F);  // qf/kf/vf/gf  |  OF/OS (aliased, lifetimes disjoint)
  const size_t oXB  = take(SZ_XD_B);
  const size_t oQB  = take(SZ_XD_B);
  const size_t oKT  = take(SZ_XD_B);
  const size_t oVT  = take(SZ_XD_B);
  const size_t oKTS = take(SZ_XD_B);
  const size_t oVTS = take(SZ_XD_B);
  const size_t oOB  = take(SZ_XD_B);
  const size_t oHB  = take(SZ_XF_B);
  const size_t oWQ  = take((size_t)NLc * Dc * Dc * 2);
  const size_t oWK  = take((size_t)NLc * Dc * Dc * 2);
  const size_t oWV  = take((size_t)NLc * Dc * Dc * 2);
  const size_t oWO  = take((size_t)NLc * Dc * Dc * 2);
  const size_t oWG  = take((size_t)NLc * Dc * Dc * 2);
  const size_t oW1  = take((size_t)NLc * FFc * Dc * 2);
  const size_t oW2  = take((size_t)NLc * Dc * FFc * 2);
  const size_t oWF  = take((size_t)NLc * 2 * sizeof(float));

  float*  X   = (float*)(ws + oX);
  float*  TF  = (float*)(ws + oTF);
  float*  QF  = (float*)(ws + oBIG);
  float*  KF  = QF + BLc * Dc;
  float*  VF  = KF + BLc * Dc;
  float*  GF  = VF + BLc * Dc;
  float*  OFp = (float*)(ws + oBIG);          // aliases QF region (dead by then)
  float*  OSp = OFp + BLc * Dc;
  bf16_t* XB  = (bf16_t*)(ws + oXB);
  bf16_t* QBp = (bf16_t*)(ws + oQB);
  bf16_t* KTp = (bf16_t*)(ws + oKT);
  bf16_t* VTp = (bf16_t*)(ws + oVT);
  bf16_t* KSp = (bf16_t*)(ws + oKTS);
  bf16_t* VSp = (bf16_t*)(ws + oVTS);
  bf16_t* OBp = (bf16_t*)(ws + oOB);
  bf16_t* HBp = (bf16_t*)(ws + oHB);
  bf16_t* WQ  = (bf16_t*)(ws + oWQ);
  bf16_t* WK  = (bf16_t*)(ws + oWK);
  bf16_t* WV  = (bf16_t*)(ws + oWV);
  bf16_t* WO  = (bf16_t*)(ws + oWO);
  bf16_t* WG  = (bf16_t*)(ws + oWG);
  bf16_t* W1p = (bf16_t*)(ws + oW1);
  bf16_t* W2p = (bf16_t*)(ws + oW2);
  float*  WFC = (float*)(ws + oWF);

  const dim3 blk32(32), blk128(128), blk256(256);
  auto cvt = [&](const float* s, bf16_t* d, size_t n) {
    cvt_bf16_kernel<<<dim3((unsigned)((n + 255) / 256)), blk256, 0, stream>>>(s, d, n);
  };
  cvt(qW, WQ, (size_t)NLc * Dc * Dc);
  cvt(kW, WK, (size_t)NLc * Dc * Dc);
  cvt(vW, WV, (size_t)NLc * Dc * Dc);
  cvt(oW, WO, (size_t)NLc * Dc * Dc);
  cvt(gW, WG, (size_t)NLc * Dc * Dc);
  cvt(fW1, W1p, (size_t)NLc * FFc * Dc);
  cvt(fW2, W2p, (size_t)NLc * Dc * FFc);

  wfac_kernel<<<dim3(NLc), blk128, 0, stream>>>(fdec, sdec, WFC);
  embed_ln_kernel<<<dim3((unsigned)BLc), blk128, 0, stream>>>(ids, we, pe, te, elnw, elnb, X);

  const dim3 gD((unsigned)(BLc / 128), Dc / 64);
  const dim3 gFF((unsigned)(BLc / 128), FFc / 64);
  const unsigned nElemD = (unsigned)(BLc * Dc / 256);

  for (int layer = 0; layer < NLc; ++layer) {
    conv_silu_kernel<<<dim3(nElemD), blk256, 0, stream>>>(
        X, convW + (size_t)layer * Dc * 4, convB + (size_t)layer * Dc, XB);

    gemm_bf16_kernel<0><<<gD, blk128, 0, stream>>>(XB, WQ + (size_t)layer * Dc * Dc, nullptr, QF, (int)BLc, Dc, Dc);
    gemm_bf16_kernel<0><<<gD, blk128, 0, stream>>>(XB, WK + (size_t)layer * Dc * Dc, nullptr, KF, (int)BLc, Dc, Dc);
    gemm_bf16_kernel<0><<<gD, blk128, 0, stream>>>(XB, WV + (size_t)layer * Dc * Dc, nullptr, VF, (int)BLc, Dc, Dc);
    gemm_bf16_kernel<0><<<gD, blk128, 0, stream>>>(XB, WG + (size_t)layer * Dc * Dc, nullptr, GF, (int)BLc, Dc, Dc);

    fuse_qkvg_kernel<<<dim3((unsigned)(BLc * Hc / 4)), blk128, 0, stream>>>(
        QF, KF, VF, GF, surW + (size_t)layer * 2 * DHc, surB + layer,
        QBp, KTp, VTp, KSp, VSp);

    delta_kernel<<<dim3((unsigned)BHc, 2), blk32, 0, stream>>>(
        QBp, KTp, VTp, KSp, VSp, WFC + layer * 2, OFp, OSp);

    blend_kernel<<<dim3(nElemD), blk256, 0, stream>>>(OFp, OSp, alpha + layer, OBp);
    gemm_bf16_kernel<0><<<gD, blk128, 0, stream>>>(OBp, WO + (size_t)layer * Dc * Dc, nullptr, TF, (int)BLc, Dc, Dc);

    residual_ln_kernel<<<dim3((unsigned)BLc), blk128, 0, stream>>>(
        X, TF, ln1w + (size_t)layer * Dc, ln1b + (size_t)layer * Dc, XB);

    // FFN1: bias + exact GeLU fused in epilogue, bf16 output (no f32 hidden round-trip)
    gemm_bf16_kernel<1><<<gFF, blk128, 0, stream>>>(XB, W1p + (size_t)layer * FFc * Dc,
                                                    fB1 + (size_t)layer * FFc, HBp, (int)BLc, FFc, Dc);
    gemm_bf16_kernel<0><<<gD, blk128, 0, stream>>>(HBp, W2p + (size_t)layer * Dc * FFc,
                                                   fB2 + (size_t)layer * Dc, TF, (int)BLc, Dc, FFc);

    residual_ln_kernel<<<dim3((unsigned)BLc), blk128, 0, stream>>>(
        X, TF, ln2w + (size_t)layer * Dc, ln2b + (size_t)layer * Dc, XB);
  }

  pool_norm_kernel<<<dim3(Bc), blk128, 0, stream>>>(X, amask, out);
}
